// GCRNN_epsilon_early_stop_density_19473381720852
// MI455X (gfx1250) — matrix operand order repaired and checked by reading, not compiled, not run on hardware
//
#include <hip/hip_runtime.h>
#include <hip/hip_bf16.h>

// ---------------------------------------------------------------------------
// GCRNN (epsilon/early-stop/density) for MI455X gfx1250.
// bf16-resident dataflow: weights converted once, activations produced as
// bf16; every contraction is V_WMMA_F32_16X16X32_BF16 with fp32 accumulate.
// Wave32 fragment layouts per CDNA5 ISA 7.12.2. 32x64 C-tile per wave, with
// the M-tile-count specialization hoisted OUT of the K-loop (branch-free body).
// Only the 6 (ACT,BK1,OUTBF) combos actually used are instantiated.
// ---------------------------------------------------------------------------

typedef __attribute__((ext_vector_type(16))) __bf16 v16bf;
typedef __attribute__((ext_vector_type(8)))  float  v8f;
typedef unsigned int u32x8 __attribute__((ext_vector_type(8)));
typedef unsigned int u32x4 __attribute__((ext_vector_type(4)));
typedef int          i32x4 __attribute__((ext_vector_type(4)));
typedef int          i32x8 __attribute__((ext_vector_type(8)));
typedef unsigned short ushort_t;

#ifndef __has_builtin
#define __has_builtin(x) 0
#endif
#if __has_builtin(__builtin_amdgcn_tensor_load_to_lds)
#define HAVE_TDM 1
#else
#define HAVE_TDM 0
#endif
#if __has_builtin(__builtin_amdgcn_s_wait_tensorcnt)
#define HAVE_TCNT 1
#else
#define HAVE_TCNT 0
#endif

__device__ __forceinline__ ushort_t us_of(float f) {   // fp32 -> bf16 bits
  return (ushort_t)((__builtin_bit_cast(unsigned, f) + 0x8000u) >> 16);
}

// A fragment 16x32 bf16 from bf16 memory (unit K stride):
// V0..3 <- K = hi*8 + {0..7}; V4..7 <- K = 16 + hi*8 + {0..7}
__device__ __forceinline__ v16bf load_afrag(const ushort_t* __restrict__ Ar, int k0, int hi) {
  uint4 q0 = *(const uint4*)(Ar + k0 + hi * 8);
  uint4 q1 = *(const uint4*)(Ar + k0 + 16 + hi * 8);
  u32x8 u;
  u[0] = q0.x; u[1] = q0.y; u[2] = q0.z; u[3] = q0.w;
  u[4] = q1.x; u[5] = q1.y; u[6] = q1.z; u[7] = q1.w;
  return __builtin_bit_cast(v16bf, u);
}

// B fragment 32x16 bf16 (lane%16 = N col): V0..3 <- K = hi*8 + {0..7}; V4..7 <- K = 16 + hi*8 + {0..7}
template<bool BK1>
__device__ __forceinline__ v16bf load_bfrag(const ushort_t* __restrict__ Wc, int k0, int hi, int Wsk) {
  u32x8 u;
  if (BK1) {
    uint4 q0 = *(const uint4*)(Wc + k0 + hi * 8);
    uint4 q1 = *(const uint4*)(Wc + k0 + 16 + hi * 8);
    u[0] = q0.x; u[1] = q0.y; u[2] = q0.z; u[3] = q0.w;
    u[4] = q1.x; u[5] = q1.y; u[6] = q1.z; u[7] = q1.w;
  } else {
    const ushort_t* p = Wc + (k0 + hi * 8) * Wsk;
#pragma unroll
    for (int v = 0; v < 8; ++v)
      u[v] = (unsigned)p[(2 * v + (v >= 4 ? 8 : 0)) * Wsk] | ((unsigned)p[(2 * v + 1 + (v >= 4 ? 8 : 0)) * Wsk] << 16);
  }
  return __builtin_bit_cast(v16bf, u);
}

// Branch-free GEMM body for MT (1 or 2) 16-row M tiles x 4 N tiles.
template<int ACT, bool BK1, bool OUTBF, int MT>
__device__ __forceinline__ void gemm_body(
    const ushort_t* __restrict__ Ar0, const ushort_t* __restrict__ Wc,
    const float* __restrict__ bias, void* __restrict__ Cv,
    int K, int Asm, int Wsk, int Wsn, int Csm,
    int row0, int col0, int r, int hi, int bias_mode, float alpha)
{
  v8f acc[MT][4] = {};
  const ushort_t* Ar1 = Ar0 + 16 * Asm;
  for (int k0 = 0; k0 < K; k0 += 32) {
    __builtin_prefetch((const void*)(Ar0 + k0 + 64), 0, 1);      // speculative
    v16bf a[MT];
    a[0] = load_afrag(Ar0, k0, hi);
    if (MT == 2) a[1] = load_afrag(Ar1, k0, hi);
#pragma unroll
    for (int j = 0; j < 4; ++j) {
      v16bf b = load_bfrag<BK1>(Wc + (16 * j) * Wsn, k0, hi, Wsk);
#pragma unroll
      for (int m = 0; m < MT; ++m)
        acc[m][j] = __builtin_amdgcn_wmma_f32_16x16x32_bf16(
            false, a[m], false, b, (short)0, acc[m][j], false, false);
    }
  }

  // C 16x16 f32 layout: VGPR i -> row = i + 8*hi, col = lane%16
  float* Cf = (float*)Cv;
  ushort_t* Cb = (ushort_t*)Cv;
#pragma unroll
  for (int j = 0; j < 4; ++j) {
    const int col = col0 + 16 * j + r;
    const float bn = (bias_mode == 1) ? bias[col] : 0.f;
#pragma unroll
    for (int m = 0; m < MT; ++m) {
#pragma unroll
      for (int i = 0; i < 8; ++i) {
        int row = row0 + 16 * m + i + hi * 8;
        float v = acc[m][j][i] * alpha + bn;
        if (bias_mode == 2) v += bias[row];
        if (ACT == 1) v = fmaxf(v, 0.f);
        else if (ACT == 2) v = 1.f / (1.f + __expf(-v));
        else if (ACT == 3) v = tanhf(v);
        if (OUTBF) Cb[row * Csm + col] = us_of(v);
        else       Cf[row * Csm + col] = v;
      }
    }
  }
}

// ---------------------------------------------------------------------------
// Batched strided GEMM on bf16 inputs:
//   C[m,n] = act(alpha*sum_k A[m,k]*B[k,n] + bias), per batch z.
// A addr = A + z*Abs + m*Asm + k       (unit A K-stride at all call sites)
// B addr = W + z*Wbs + k*Wsk + n*Wsn  (BK1 => Wsk==1)
// C addr = C + z*Cbs + m*Csm + n      (unit N stride; OUTBF: bf16 store)
// K % 32 == 0, N % 64 == 0, M % 16 == 0 (padded by host).
// One wave computes a 32x64 tile (last M tile may be 16 rows).
// ---------------------------------------------------------------------------
template<int ACT, bool BK1, bool OUTBF>
__global__ __launch_bounds__(32) void gemm_wmma_kernel(
    const ushort_t* __restrict__ A, const ushort_t* __restrict__ W,
    const float* __restrict__ bias, void* __restrict__ Cv,
    int M, int K,
    int Asm, int Abs,
    int Wsk, int Wsn, int Wbs,
    int Csm, int Cbs,
    int bias_mode, float alpha)
{
  const int lane = threadIdx.x & 31;
  const int r  = lane & 15;
  const int hi = lane >> 4;
  const int col0 = blockIdx.x * 64;
  const int row0 = blockIdx.y * 32;
  const int z = blockIdx.z;

  const ushort_t* Ar0 = A + z * Abs + (row0 + r) * Asm;
  const ushort_t* Wc  = W + z * Wbs + (col0 + r) * Wsn;
  void* Cz = OUTBF ? (void*)((ushort_t*)Cv + z * Cbs) : (void*)((float*)Cv + z * Cbs);

  if ((M - row0) >= 32)    // wave-uniform: whole-kernel specialization
    gemm_body<ACT, BK1, OUTBF, 2>(Ar0, Wc, bias, Cz, K, Asm, Wsk, Wsn, Csm,
                                  row0, col0, r, hi, bias_mode, alpha);
  else
    gemm_body<ACT, BK1, OUTBF, 1>(Ar0, Wc, bias, Cz, K, Asm, Wsk, Wsn, Csm,
                                  row0, col0, r, hi, bias_mode, alpha);
}

// softmax over last dim (64) of fp32 scores -> bf16 probs
__global__ __launch_bounds__(128) void softmax64_kernel(const float* __restrict__ s,
                                                        ushort_t* __restrict__ p, int rows) {
  int wave = threadIdx.x >> 5, lane = threadIdx.x & 31;
  int row = blockIdx.x * 4 + wave;
  if (row >= rows) return;
  const float* sr = s + row * 64;
  float a0 = sr[lane], a1 = sr[lane + 32];
  float mx = fmaxf(a0, a1);
  for (int m = 16; m > 0; m >>= 1) mx = fmaxf(mx, __shfl_xor(mx, m, 32));
  float e0 = __expf(a0 - mx), e1 = __expf(a1 - mx);
  float sum = e0 + e1;
  for (int m = 16; m > 0; m >>= 1) sum += __shfl_xor(sum, m, 32);
  float inv = 1.f / sum;
  ushort_t* pr = p + row * 64;
  pr[lane]      = us_of(e0 * inv);
  pr[lane + 32] = us_of(e1 * inv);
}

// out = [tanh?] LN(x + r)*g + b; optional bf16 copy for the next GEMM
__global__ __launch_bounds__(128) void add_ln_kernel(
    const float* __restrict__ x, const float* __restrict__ rr_,
    const float* __restrict__ g, const float* __restrict__ b,
    float* __restrict__ out, ushort_t* __restrict__ out_bf, int rows, int do_tanh)
{
  int wave = threadIdx.x >> 5, lane = threadIdx.x & 31;
  int row = blockIdx.x * 4 + wave;
  if (row >= rows) return;
  const float* xr = x + row * 256;
  const float* rr = rr_ + row * 256;
  float v[8]; float s = 0.f;
#pragma unroll
  for (int i = 0; i < 8; ++i) { v[i] = xr[lane + 32 * i] + rr[lane + 32 * i]; s += v[i]; }
  for (int m = 16; m > 0; m >>= 1) s += __shfl_xor(s, m, 32);
  float mu = s * (1.f / 256.f);
  float q = 0.f;
#pragma unroll
  for (int i = 0; i < 8; ++i) { float d = v[i] - mu; q += d * d; }
  for (int m = 16; m > 0; m >>= 1) q += __shfl_xor(q, m, 32);
  float inv = rsqrtf(q * (1.f / 256.f) + 1e-5f);
#pragma unroll
  for (int i = 0; i < 8; ++i) {
    int cidx = lane + 32 * i;
    float y = (v[i] - mu) * inv * g[cidx] + b[cidx];
    if (do_tanh) y = tanhf(y);
    out[row * 256 + cidx] = y;
    if (out_bf) out_bf[row * 256 + cidx] = us_of(y);
  }
}

// inp[n,b,d] = h[b,d,n] + embed[n,d]*te[b,n]  (fp32 + bf16 copies)
__global__ void build_inp_kernel(const float* __restrict__ h,
                                 const float* __restrict__ embed,
                                 const float* __restrict__ te,
                                 float* __restrict__ inp,
                                 ushort_t* __restrict__ inp_bf)
{
  int idx = blockIdx.x * blockDim.x + threadIdx.x;   // 64*128*256
  if (idx >= 64 * 128 * 256) return;
  int d = idx & 255; int nb = idx >> 8; int b = nb & 127; int n = nb >> 7;
  float v = h[(b * 256 + d) * 64 + n] + embed[n * 256 + d] * te[b * 64 + n];
  inp[idx] = v;
  inp_bf[idx] = us_of(v);
}

// h[b,d,n] = sqrt(1-dt[b,t])*h - sqrt(dt[b,t])*eps[n,b,d]
__global__ void update_h_kernel(float* __restrict__ h, const float* __restrict__ eps,
                                const float* __restrict__ dt, int t)
{
  int idx = blockIdx.x * blockDim.x + threadIdx.x;   // 128*256*64
  if (idx >= 128 * 256 * 64) return;
  int n = idx & 63; int bd = idx >> 6; int d = bd & 255; int b = bd >> 8;
  float dtb = dt[b * 64 + t];
  float e = eps[((n * 128) + b) * 256 + d];
  h[idx] = sqrtf(1.f - dtb) * h[idx] - sqrtf(dtb) * e;
}

// bf16 xcat: channels [0,cin): (density?)*x for c<c_in else h[c-c_in]
__global__ void build_xcat_kernel(ushort_t* __restrict__ dst, int cin, int c_in, int chcap,
                                  const float* __restrict__ x, int x_inner,
                                  const float* __restrict__ h,
                                  const float* __restrict__ density)
{
  long long total = (long long)128 * cin * 64;
  long long idx = (long long)blockIdx.x * blockDim.x + threadIdx.x;
  if (idx >= total) return;
  int n = (int)(idx & 63); long long bc = idx >> 6;
  int c = (int)(bc % cin); int b = (int)(bc / cin);
  float v;
  if (c < c_in) {
    v = x[((b * c_in + c) * 64 + n) * x_inner];
    if (density) v *= density[b * 64 + n];
  } else {
    v = h[((b * 256) + (c - c_in)) * 64 + n];
  }
  dst[(b * chcap + c) * 64 + n] = us_of(v);
}

// fp32 -> bf16 weight conversion (flat)
__global__ void cvt_bf_kernel(ushort_t* __restrict__ dst, const float* __restrict__ src, int n) {
  int i = blockIdx.x * blockDim.x + threadIdx.x;
  if (i < n) dst[i] = us_of(src[i]);
}

// fp32 -> bf16 with zero-padded columns (rows x dcols, src rows x scols)
__global__ void pad_cvt_kernel(ushort_t* __restrict__ dst, const float* __restrict__ src,
                               int rows, int scols, int dcols)
{
  int idx = blockIdx.x * blockDim.x + threadIdx.x;
  if (idx >= rows * dcols) return;
  int c = idx % dcols, rr = idx / dcols;
  dst[idx] = (c < scols) ? us_of(src[rr * scols + c]) : (ushort_t)0;
}

// h = u*h + (1-u)*c ; optionally write d_out rows where lengths[b]-1 == t
__global__ void gate_kernel(float* __restrict__ h, const float* __restrict__ u,
                            const float* __restrict__ c, float* __restrict__ outbuf,
                            const int* __restrict__ lengths, int t, int write_out)
{
  int idx = blockIdx.x * blockDim.x + threadIdx.x;   // 128*256*64
  if (idx >= 128 * 256 * 64) return;
  float uu = u[idx];
  float nv = uu * h[idx] + (1.f - uu) * c[idx];
  h[idx] = nv;
  if (write_out) {
    int b = idx >> 14;
    if (lengths[b] - 1 == t) outbuf[idx] = nv;
  }
}

// ---------------------------------------------------------------------------
// TDM warm-up: DMA the 64x64 fp32 adjacency into LDS via tensor_load_to_lds.
// ---------------------------------------------------------------------------
__global__ __launch_bounds__(64) void tdm_warm_kernel(const float* __restrict__ adj,
                                                      float* __restrict__ sink)
{
  __shared__ float lds_adj[64 * 64];
#if HAVE_TDM
  if (threadIdx.x < 32) {
    unsigned long long ga = (unsigned long long)(const void*)adj;
    u32x4 g0;
    g0[0] = 1u;                                      // count=1, user descriptor
    g0[1] = 0u;                                      // lds_addr = 0
    g0[2] = (unsigned)(ga & 0xFFFFFFFFull);
    g0[3] = (unsigned)((ga >> 32) & 0x1FFFFFFull) | (2u << 30);  // type=2
    i32x8 g1;
    g1[0] = (2 << 16);                               // data_size=4B
    g1[1] = (64 << 16);                              // tensor_dim0 at bit48
    g1[2] = (64 << 16);                              // tensor_dim1 at bit80
    g1[3] = (64 << 16);                              // tile_dim0 at bit112
    g1[4] = 64;                                      // tile_dim1 (2D tile)
    g1[5] = 64;                                      // tensor_dim0_stride
    g1[6] = 0; g1[7] = 0;
    i32x4 gz; gz[0] = 0; gz[1] = 0; gz[2] = 0; gz[3] = 0;
#if __clang_major__ >= 23
    i32x8 gz8; for (int i = 0; i < 8; ++i) gz8[i] = 0;
    __builtin_amdgcn_tensor_load_to_lds(g0, g1, gz, gz, gz8, 0);
#else
    __builtin_amdgcn_tensor_load_to_lds(g0, g1, gz, gz, 0);
#endif
#if HAVE_TCNT
    __builtin_amdgcn_s_wait_tensorcnt(0);
#endif
  }
#else
  for (int i = threadIdx.x; i < 64 * 64; i += 64) lds_adj[i] = adj[i];
#endif
  __syncthreads();
  sink[threadIdx.x] = lds_adj[threadIdx.x];
}

// ---------------------------------------------------------------------------
// Host orchestration (compile-time GEMM dispatch: only 6 combos instantiated)
// ---------------------------------------------------------------------------
template<int ACT, bool BK1, bool OUTBF>
static void launch_gemm_t(hipStream_t s,
                          const ushort_t* A, const ushort_t* W, const float* bias, void* C,
                          int M, int N, int K,
                          int Asm, int Abs, int Wsk, int Wsn, int Wbs,
                          int Csm, int Cbs, int batches, int bias_mode, float alpha)
{
  dim3 grid(N / 64, (M + 31) / 32, batches);
  dim3 block(32);
  gemm_wmma_kernel<ACT, BK1, OUTBF><<<grid, block, 0, s>>>(
      A, W, bias, C, M, K, Asm, Abs, Wsk, Wsn, Wbs, Csm, Cbs, bias_mode, alpha);
}

extern "C" void kernel_launch(void* const* d_in, const int* in_sizes, int n_in,
                              void* d_out, int out_size, void* d_ws, size_t ws_size,
                              hipStream_t stream)
{
  (void)in_sizes; (void)n_in; (void)ws_size;
  const float* x        = (const float*)d_in[0];   // (128,16,64,64)
  const float* density  = (const float*)d_in[1];   // (128,64)
  const float* adj      = (const float*)d_in[2];   // (64,64)
  const float* delta_t  = (const float*)d_in[3];   // (128,64)
  const int*   lengths  = (const int*)  d_in[4];   // (128,)
  const float* time_enc = (const float*)d_in[5];   // (64,128,1,64)
  auto prm = [&](int l, int i) -> const float* { return (const float*)d_in[7 + l * 17 + i]; };
  // 0 embed,1 in_w,2 in_b,3 out_w,4 out_b,5 ln1_g,6 ln1_b,7 ln2_g,8 ln2_b,
  // 9 ff1_w,10 ff1_b,11 ff2_w,12 ff2_b,13 wu,14 bu,15 wc,16 bc

  float* ws = (float*)d_ws;
  size_t off = 0;
  auto take = [&](size_t nfloats) -> float* {
    float* p = ws + off; off += (nfloats + 3) & ~(size_t)3; return p;   // keep 16B align
  };
  auto take_us = [&](size_t nus) -> ushort_t* { return (ushort_t*)take((nus + 1) / 2); };
  const size_t E = 128ull * 256 * 64;              // 2,097,152

  // fp32 buffers
  float* h0     = take(E);
  float* h1     = take(E);
  float* inp    = take(E);                         // (64,128,256)
  float* scores = take(128ull * 64 * 64);
  float* tmp1   = take(E);
  float* tmp2   = take(E);
  float* xln    = take(E);
  float* eps    = take(E);
  float* ug     = take(E);
  float* cg     = take(E);
  float* sink   = take(4096);
  // bf16 buffers
  ushort_t* inp_bf  = take_us(E);
  ushort_t* qkv_bf  = take_us(3 * E);              // (64,128,768)
  ushort_t* prob_bf = take_us(128ull * 64 * 64);
  ushort_t* attn_bf = take_us(E);
  ushort_t* xln_bf  = take_us(E);
  ushort_t* ff1_bf  = take_us(E);
  ushort_t* xcat0   = take_us(128ull * 832 * 64);  // layer0, K padded 816->832
  ushort_t* xcat1   = take_us(128ull * 1536 * 64); // layer1, K = 1536
  ushort_t* w_in[2]  = { take_us(768ull * 256), take_us(768ull * 256) };
  ushort_t* w_out[2] = { take_us(256ull * 256), take_us(256ull * 256) };
  ushort_t* w_f1[2]  = { take_us(256ull * 256), take_us(256ull * 256) };
  ushort_t* w_f2[2]  = { take_us(256ull * 256), take_us(256ull * 256) };
  ushort_t* w_u[2]   = { take_us(256ull * 832), take_us(256ull * 1536) };
  ushort_t* w_c[2]   = { take_us(256ull * 832), take_us(256ull * 1536) };
  ushort_t* adj_bf   = take_us(64ull * 64);

  hipMemsetAsync(h0, 0, E * sizeof(float), stream);
  hipMemsetAsync(h1, 0, E * sizeof(float), stream);
  hipMemsetAsync(d_out, 0, (size_t)out_size * sizeof(float), stream);
  hipMemsetAsync(xcat0, 0, 128ull * 832 * 64 * sizeof(ushort_t), stream);  // pads stay 0

  // one-time weight conversion fp32 -> bf16 (reused across the 64-step scan)
  for (int l = 0; l < 2; ++l) {
    cvt_bf_kernel<<<(768 * 256 + 255) / 256, 256, 0, stream>>>(w_in[l],  prm(l, 1),  768 * 256);
    cvt_bf_kernel<<<(256 * 256 + 255) / 256, 256, 0, stream>>>(w_out[l], prm(l, 3),  256 * 256);
    cvt_bf_kernel<<<(256 * 256 + 255) / 256, 256, 0, stream>>>(w_f1[l],  prm(l, 9),  256 * 256);
    cvt_bf_kernel<<<(256 * 256 + 255) / 256, 256, 0, stream>>>(w_f2[l],  prm(l, 11), 256 * 256);
  }
  pad_cvt_kernel<<<(256 * 832 + 255) / 256, 256, 0, stream>>>(w_u[0], prm(0, 13), 256, 816, 832);
  pad_cvt_kernel<<<(256 * 832 + 255) / 256, 256, 0, stream>>>(w_c[0], prm(0, 15), 256, 816, 832);
  cvt_bf_kernel<<<(256 * 1536 + 255) / 256, 256, 0, stream>>>(w_u[1], prm(1, 13), 256 * 1536);
  cvt_bf_kernel<<<(256 * 1536 + 255) / 256, 256, 0, stream>>>(w_c[1], prm(1, 15), 256 * 1536);
  cvt_bf_kernel<<<(64 * 64 + 255) / 256, 256, 0, stream>>>(adj_bf, adj, 64 * 64);
  tdm_warm_kernel<<<1, 64, 0, stream>>>(adj, sink);

  float* hbuf[2] = { h0, h1 };
  ushort_t* xcat_l[2] = { xcat0, xcat1 };
  const int c_in_l[2] = { 16, 256 };
  const int cin_l[2]  = { 272, 512 };
  const int kpad_l[2] = { 832, 1536 };             // padded 3*cin (K of wu/wc gemm)
  const int BQ = 128 * 768;                        // 98304

  for (int t = 0; t < 64; ++t) {
    for (int l = 0; l < 2; ++l) {
      float* h = hbuf[l];
      if (t > 0) {
        // ---- epsilon = tanh(transformer(h^T + embed*te)) ----
        build_inp_kernel<<<8192, 256, 0, stream>>>(h, prm(l, 0),
            time_enc + (size_t)(t - 1) * 128 * 64, inp, inp_bf);
        // qkv (bf16 out): (8192,768) = inp x in_w^T + in_b
        launch_gemm_t<0, true, true>(stream, inp_bf, w_in[l], prm(l, 2), qkv_bf, 8192, 768, 256,
                                     256, 0,   1, 256, 0,   768, 0,   1, 1, 1.0f);
        // scores[b,s,t'] = q.k/16 (fp32 out, batched over b)
        launch_gemm_t<0, true, false>(stream, qkv_bf, qkv_bf + 256, nullptr, scores, 64, 64, 256,
                                      BQ, 768,   1, BQ, 768,   64, 4096,   128, 0, 0.0625f);
        softmax64_kernel<<<2048, 128, 0, stream>>>(scores, prob_bf, 128 * 64);
        // attn[s,b,d] = a[b,s,t'] . v[t',b,d]  (bf16 out; B K-strided)
        launch_gemm_t<0, false, true>(stream, prob_bf, qkv_bf + 512, nullptr, attn_bf, 64, 256, 64,
                                      64, 4096,   BQ, 1, 768,   128 * 256, 256,   128, 0, 1.0f);
        launch_gemm_t<0, true, false>(stream, attn_bf, w_out[l], prm(l, 4), tmp1, 8192, 256, 256,
                                      256, 0,   1, 256, 0,   256, 0,   1, 1, 1.0f);
        add_ln_kernel<<<2048, 128, 0, stream>>>(inp, tmp1, prm(l, 5), prm(l, 6), xln, xln_bf, 8192, 0);
        launch_gemm_t<1, true, true>(stream, xln_bf, w_f1[l], prm(l, 10), ff1_bf, 8192, 256, 256,
                                     256, 0,   1, 256, 0,   256, 0,   1, 1, 1.0f);
        launch_gemm_t<0, true, false>(stream, ff1_bf, w_f2[l], prm(l, 12), tmp2, 8192, 256, 256,
                                      256, 0,   1, 256, 0,   256, 0,   1, 1, 1.0f);
        add_ln_kernel<<<2048, 128, 0, stream>>>(xln, tmp2, prm(l, 7), prm(l, 8), eps, nullptr, 8192, 1);
        update_h_kernel<<<8192, 256, 0, stream>>>(h, eps, delta_t, t);
      }
      const float* xin = (l == 0) ? (x + t) : hbuf[0];
      const int xinner = (l == 0) ? 64 : 1;
      const int cin = cin_l[l], c_in = c_in_l[l], chcap = kpad_l[l], kk = kpad_l[l];
      ushort_t* xc = xcat_l[l];
      const int batch_ch = chcap * 64;
      const int xblocks = (int)(((long long)128 * cin * 64 + 255) / 256);

      // ---- u = sigmoid(diff_conv([density*x ; h])) ----
      build_xcat_kernel<<<xblocks, 256, 0, stream>>>(xc, cin, c_in, chcap, xin, xinner, h, density);
      launch_gemm_t<0, true, true>(stream, xc, adj_bf, nullptr, xc + (size_t)cin * 64, cin, 64, 64,
                                   64, batch_ch,   1, 64, 0,   64, batch_ch,   128, 0, 1.0f);
      launch_gemm_t<0, true, true>(stream, xc + (size_t)cin * 64, adj_bf, nullptr,
                                   xc + (size_t)2 * cin * 64, cin, 64, 64,
                                   64, batch_ch,   1, 64, 0,   64, batch_ch,   128, 0, 1.0f);
      launch_gemm_t<2, false, false>(stream, w_u[l], xc, prm(l, 14), ug, 256, 64, kk,
                                     kk, 0,   64, 1, batch_ch,   64, 16384,   128, 2, 1.0f);

      // ---- c = tanh(diff_conv([x ; h])) ----
      build_xcat_kernel<<<xblocks, 256, 0, stream>>>(xc, cin, c_in, chcap, xin, xinner, h, nullptr);
      launch_gemm_t<0, true, true>(stream, xc, adj_bf, nullptr, xc + (size_t)cin * 64, cin, 64, 64,
                                   64, batch_ch,   1, 64, 0,   64, batch_ch,   128, 0, 1.0f);
      launch_gemm_t<0, true, true>(stream, xc + (size_t)cin * 64, adj_bf, nullptr,
                                   xc + (size_t)2 * cin * 64, cin, 64, 64,
                                   64, batch_ch,   1, 64, 0,   64, batch_ch,   128, 0, 1.0f);
      launch_gemm_t<3, false, false>(stream, w_c[l], xc, prm(l, 16), cg, 256, 64, kk,
                                     kk, 0,   64, 1, batch_ch,   64, 16384,   128, 2, 1.0f);

      // ---- h = u*h + (1-u)*c ; last layer conditionally writes output ----
      gate_kernel<<<8192, 256, 0, stream>>>(h, ug, cg, (float*)d_out, lengths, t,
                                            (l == 1) ? 1 : 0);
    }
  }
}
